// DecCLSTMBlock_91233695301776
// MI455X (gfx1250) — matrix operand order repaired and checked by reading, not compiled, not run on hardware
//
#include <hip/hip_runtime.h>

// ---------------------------------------------------------------------------
// DecCLSTMBlock for MI455X (gfx1250): 3x ConvLSTM2D + BN + 2x2 upsample.
// Convs as implicit GEMM on V_WMMA_F32_16X16X32_BF16 (bf16 in, f32 acc).
// Halo-padded feature maps -> no bounds checks in the GEMM loop.
// 2x2 register blocking: 8x global_load_b128 feed 4x v_wmma per K-group.
// ---------------------------------------------------------------------------

typedef __attribute__((ext_vector_type(16))) __bf16 v16bf;
typedef __attribute__((ext_vector_type(8)))  float  v8f;
typedef __attribute__((ext_vector_type(8)))  int    v8i;
typedef __attribute__((ext_vector_type(4)))  int    v4i;

#define B_ 2
#define T_ 8

__device__ __forceinline__ unsigned short f2bf(float f) {
  // round-to-nearest-even fp32 -> bf16
  unsigned u = __builtin_bit_cast(unsigned, f);
  unsigned r = u + 0x7FFFu + ((u >> 16) & 1u);
  return (unsigned short)(r >> 16);
}

__device__ __forceinline__ float hsig(float x) {
  return fminf(fmaxf(0.2f * x + 0.5f, 0.0f), 1.0f);
}

// Load a 16-element bf16 fragment as two contiguous b128 runs (run2 at +off
// shorts), matching the CDNA5 A/B per-lane packing.
__device__ __forceinline__ v16bf load_frag(const unsigned short* base, int off) {
  const v4i lo = *(const v4i*)(base);
  const v4i hi = *(const v4i*)(base + off);
  v8i v;
#pragma unroll
  for (int i = 0; i < 4; ++i) { v[i] = lo[i]; v[i + 4] = hi[i]; }
  return __builtin_bit_cast(v16bf, v);
}

// ---------------------------------------------------------------------------
// Weight prep: Wb[kk][n][c] = (c<Cin ? Wx[kk][c][n] : Wh[kk][c-Cin][n]) as bf16
// K (=c) contiguous so b128 loads grab 8 packed bf16 pairs for B fragments.
// ---------------------------------------------------------------------------
__global__ void prep_w_kernel(const float* __restrict__ Wx,
                              const float* __restrict__ Wh,
                              unsigned short* __restrict__ Wb,
                              int Cin, int Co4, int Ctot, int total) {
  int idx = blockIdx.x * blockDim.x + threadIdx.x;
  if (idx >= total) return;
  int kk  = idx / (Co4 * Ctot);
  int rem = idx - kk * (Co4 * Ctot);
  int n   = rem / Ctot;
  int c   = rem - n * Ctot;
  float v;
  if (c < Cin) v = Wx[((size_t)(kk * Cin + c)) * Co4 + n];
  else         v = Wh[((size_t)(kk * (Ctot - Cin) + (c - Cin))) * Co4 + n];
  Wb[idx] = f2bf(v);
}

// ---------------------------------------------------------------------------
// Zero the whole halo-padded feature buffer (halo stays zero for all t;
// interior zeros double as h_{-1} = 0).
// ---------------------------------------------------------------------------
__global__ void zero_f_kernel(unsigned short* __restrict__ F, int total) {
  int idx = blockIdx.x * blockDim.x + threadIdx.x;
  if (idx < total) F[idx] = 0;
}

// ---------------------------------------------------------------------------
// Pack the x-slice into the interior of F[B][H+2][W+2][Ctot] (bf16), with the
// 2x2 upsample of the previous layer's BN output folded in.
// ---------------------------------------------------------------------------
__global__ void pack_x_kernel(const float* __restrict__ src,
                              unsigned short* __restrict__ F,
                              int H, int W, int Cin, int Ctot,
                              int t, int srcH, int srcW, int upsamp,
                              int total) {
  int idx = blockIdx.x * blockDim.x + threadIdx.x;
  if (idx >= total) return;  // total = B*H*W*Cin
  int c = idx % Cin;
  int p = idx / Cin;
  int HW = H * W;
  int b = p / HW;
  int rem = p - b * HW;
  int y = rem / W;
  int x = rem - y * W;
  int ys = upsamp ? (y >> 1) : y;
  int xs = upsamp ? (x >> 1) : x;
  float v = src[((((size_t)b * T_ + t) * srcH + ys) * srcW + xs) * Cin + c];
  F[((size_t)(b * (H + 2) + y + 1) * (W + 2) + x + 1) * Ctot + c] = f2bf(v);
}

// ---------------------------------------------------------------------------
// Implicit-GEMM conv on halo-padded F, 2x2 register-blocked:
// one wave -> 32 pixels x 32 gate channels (4 f32 accumulators).
// Fragment layouts per CDNA5 ISA 7.12.2 (wave32):
//   A (16x32 bf16): lane&15 = M; lane-half K-offset 0/8;
//                   pairs at K = khalfA+[0..7] and khalfA+16+[0..7] (2x 16B).
//   B (32x16 bf16): lane&15 = N; lane-half K-offset 0/8;
//                   pairs at K = khalfA+[0..7] and khalfA+16+[0..7] (2x 16B).
//   C/D (16x16 f32): VGPR r -> M = r + 8*(lane>=16), N = lane&15.
// ---------------------------------------------------------------------------
__global__ __launch_bounds__(256) void gemm_z_kernel(
    const unsigned short* __restrict__ F,   // [B][(H+2)][(W+2)][Ctot] bf16
    const unsigned short* __restrict__ Wb,  // [9][Co4][Ctot] bf16
    const float* __restrict__ bias,         // [Co4]
    float* __restrict__ Z,                  // [B*H*W][Co4]
    int H, int W, int Ctot, int Co4) {
  const int lane = threadIdx.x & 31;
  const int wave = threadIdx.x >> 5;
  const int p0 = blockIdx.x * 32;                // 32-pixel macro tile
  const int n0 = (blockIdx.y * 8 + wave) * 32;   // 32-channel macro tile
  if (n0 >= Co4) return;                         // wave-uniform guard

  const int khalfA = (lane >> 4) << 3;  // 0 or 8
  const int khalfB = (lane >> 4) << 4;  // 0 or 16
  const int Wp = W + 2;
  const int Hp = H + 2;
  const int HW = H * W;

  // two pixel sub-tiles: p0+[0..15], p0+16+[0..15]
  const int pA = p0 + (lane & 15);
  const int pB = pA + 16;
  const int bimgA = pA / HW;
  const int remA = pA - bimgA * HW;
  const int yA = remA / W, xA = remA - (remA / W) * W;
  const int bimgB = pB / HW;
  const int remB = pB - bimgB * HW;
  const int yB = remB / W, xB = remB - (remB / W) * W;

  // two channel sub-tiles: n0+[0..15], n0+16+[0..15]
  const int nl0 = n0 + (lane & 15);
  const int nl1 = nl0 + 16;

  v8f acc00, acc01, acc10, acc11;  // [pixelTile][chanTile]
  const float bv0 = bias[nl0];
  const float bv1 = bias[nl1];
#pragma unroll
  for (int r = 0; r < 8; ++r) {
    acc00[r] = bv0; acc01[r] = bv1;
    acc10[r] = bv0; acc11[r] = bv1;
  }

  for (int ky = 0; ky < 3; ++ky) {
    for (int kx = 0; kx < 3; ++kx) {
      const unsigned short* ArowA =
          F + ((size_t)(bimgA * Hp + yA + ky) * Wp + xA + kx) * Ctot + khalfA;
      const unsigned short* ArowB =
          F + ((size_t)(bimgB * Hp + yB + ky) * Wp + xB + kx) * Ctot + khalfA;
      const size_t wrow = (size_t)((ky * 3 + kx) * Co4);
      const unsigned short* Brow0 = Wb + (wrow + nl0) * Ctot + khalfA;
      const unsigned short* Brow1 = Wb + (wrow + nl1) * Ctot + khalfA;
      for (int cg = 0; cg < Ctot; cg += 32) {
        const v16bf fa0 = load_frag(ArowA + cg, 16);
        const v16bf fa1 = load_frag(ArowB + cg, 16);
        const v16bf fb0 = load_frag(Brow0 + cg, 16);
        const v16bf fb1 = load_frag(Brow1 + cg, 16);
        acc00 = __builtin_amdgcn_wmma_f32_16x16x32_bf16(
            false, fa0, false, fb0, (short)0, acc00, false, false);
        acc01 = __builtin_amdgcn_wmma_f32_16x16x32_bf16(
            false, fa0, false, fb1, (short)0, acc01, false, false);
        acc10 = __builtin_amdgcn_wmma_f32_16x16x32_bf16(
            false, fa1, false, fb0, (short)0, acc10, false, false);
        acc11 = __builtin_amdgcn_wmma_f32_16x16x32_bf16(
            false, fa1, false, fb1, (short)0, acc11, false, false);
      }
    }
  }

#pragma unroll
  for (int r = 0; r < 8; ++r) {
    const int mr = r + khalfA;  // C/D layout: M = r + 8*(lane>=16)
    Z[(size_t)(p0 + mr) * Co4 + nl0]      = acc00[r];
    Z[(size_t)(p0 + mr) * Co4 + nl1]      = acc01[r];
    Z[(size_t)(p0 + 16 + mr) * Co4 + nl0] = acc10[r];
    Z[(size_t)(p0 + 16 + mr) * Co4 + nl1] = acc11[r];
  }
}

// ---------------------------------------------------------------------------
// Gate math (Keras i,f,g,o order) + cell/h update + BN(h) to layer output,
// h fed back (bf16) into F's interior h channels for the next timestep.
// ---------------------------------------------------------------------------
__global__ void gates_kernel(const float* __restrict__ Z,
                             float* __restrict__ Cst,
                             unsigned short* __restrict__ F,
                             float* __restrict__ Lout,
                             const float* __restrict__ gam,
                             const float* __restrict__ bet,
                             const float* __restrict__ mmean,
                             const float* __restrict__ mvar,
                             int H, int W, int Co, int Cin, int Ctot,
                             int t, int first, int total) {
  int idx = blockIdx.x * blockDim.x + threadIdx.x;
  if (idx >= total) return;
  int p = idx / Co;
  int co = idx - p * Co;
  const size_t zb = (size_t)p * (4 * Co);
  float zi = Z[zb + co];
  float zf = Z[zb + Co + co];
  float zg = Z[zb + 2 * Co + co];
  float zo = Z[zb + 3 * Co + co];
  float cold = first ? 0.0f : Cst[idx];
  float cnew = hsig(zf) * cold + hsig(zi) * tanhf(zg);
  float h = hsig(zo) * tanhf(cnew);
  Cst[idx] = cnew;
  int HW = H * W;
  int b = p / HW;
  int rem = p - b * HW;
  int y = rem / W;
  int x = rem - y * W;
  F[((size_t)(b * (H + 2) + y + 1) * (W + 2) + x + 1) * Ctot + Cin + co] = f2bf(h);
  // BN inference, eps = 1e-3
  float ho = (h - mmean[co]) * rsqrtf(mvar[co] + 1e-3f) * gam[co] + bet[co];
  Lout[((((size_t)b * T_ + t) * H + y) * W + x) * Co + co] = ho;
}

// ---------------------------------------------------------------------------
// Final 2x2 nearest upsample of layer-3 BN output into d_out.
// ---------------------------------------------------------------------------
__global__ void upsample_out_kernel(const float* __restrict__ L3,
                                    float* __restrict__ out,
                                    int H, int W, int Co, long total) {
  long idx = (long)blockIdx.x * blockDim.x + threadIdx.x;
  if (idx >= total) return;
  int c = (int)(idx % Co);
  long r = idx / Co;
  int x = (int)(r % (2 * W)); r /= (2 * W);
  int y = (int)(r % (2 * H)); r /= (2 * H);
  int t = (int)(r % T_);
  int b = (int)(r / T_);
  out[idx] = L3[((((size_t)b * T_ + t) * H + (y >> 1)) * W + (x >> 1)) * Co + c];
}

// ---------------------------------------------------------------------------
extern "C" void kernel_launch(void* const* d_in, const int* in_sizes, int n_in,
                              void* d_out, int out_size, void* d_ws, size_t ws_size,
                              hipStream_t stream) {
  (void)in_sizes; (void)n_in; (void)out_size; (void)ws_size;
  const float* x0 = (const float*)d_in[0];
  const float* Wx[3]  = {(const float*)d_in[1],  (const float*)d_in[8],  (const float*)d_in[15]};
  const float* Wh[3]  = {(const float*)d_in[2],  (const float*)d_in[9],  (const float*)d_in[16]};
  const float* bb[3]  = {(const float*)d_in[3],  (const float*)d_in[10], (const float*)d_in[17]};
  const float* gg[3]  = {(const float*)d_in[4],  (const float*)d_in[11], (const float*)d_in[18]};
  const float* be[3]  = {(const float*)d_in[5],  (const float*)d_in[12], (const float*)d_in[19]};
  const float* mme[3] = {(const float*)d_in[6],  (const float*)d_in[13], (const float*)d_in[20]};
  const float* mva[3] = {(const float*)d_in[7],  (const float*)d_in[14], (const float*)d_in[21]};

  const int Cin[3] = {192, 128, 64};
  const int Co[3]  = {128, 64, 32};
  const int Hs[3]  = {16, 32, 64};

  char* ws = (char*)d_ws;
  size_t off = 0;
  auto carve = [&](size_t bytes) -> char* {
    char* pp = ws + off;
    off = (off + bytes + 255) & ~(size_t)255;
    return pp;
  };

  unsigned short* WbP[3];
  unsigned short* FP[3];
  float *ZP[3], *CP[3], *LP[3];
  for (int l = 0; l < 3; ++l) {
    const int Ctot = Cin[l] + Co[l];
    const int Co4 = 4 * Co[l];
    const int H = Hs[l];
    const int Npix = B_ * H * H;
    const int NpixPad = B_ * (H + 2) * (H + 2);
    WbP[l] = (unsigned short*)carve((size_t)9 * Co4 * Ctot * 2);
    FP[l]  = (unsigned short*)carve((size_t)NpixPad * Ctot * 2);
    ZP[l]  = (float*)carve((size_t)Npix * Co4 * 4);
    CP[l]  = (float*)carve((size_t)Npix * Co[l] * 4);
    LP[l]  = (float*)carve((size_t)B_ * T_ * H * H * Co[l] * 4);
  }

  // 1) weight prep (once)
  for (int l = 0; l < 3; ++l) {
    const int Ctot = Cin[l] + Co[l];
    const int Co4 = 4 * Co[l];
    const int total = 9 * Co4 * Ctot;
    prep_w_kernel<<<(total + 255) / 256, 256, 0, stream>>>(
        Wx[l], Wh[l], WbP[l], Cin[l], Co4, Ctot, total);
  }

  // 2) layers x timesteps
  for (int l = 0; l < 3; ++l) {
    const int H = Hs[l], W = H;
    const int Ctot = Cin[l] + Co[l];
    const int Co4 = 4 * Co[l];
    const int Npix = B_ * H * W;
    const int ftotal = B_ * (H + 2) * (W + 2) * Ctot;
    const float* src = (l == 0) ? x0 : LP[l - 1];
    const int srcH = (l == 0) ? H : Hs[l - 1];
    const int up = (l == 0) ? 0 : 1;
    // macro tiles: 32 pixels x (8 waves * 32 channels)
    dim3 ggrid((unsigned)(Npix / 32), (unsigned)((Co4 + 255) / 256));

    // zero halo + h_{-1}
    zero_f_kernel<<<(ftotal + 255) / 256, 256, 0, stream>>>(FP[l], ftotal);

    for (int t = 0; t < T_; ++t) {
      const int ptotal = Npix * Cin[l];
      pack_x_kernel<<<(ptotal + 255) / 256, 256, 0, stream>>>(
          src, FP[l], H, W, Cin[l], Ctot, t, srcH, srcH, up, ptotal);
      gemm_z_kernel<<<ggrid, 256, 0, stream>>>(FP[l], WbP[l], bb[l], ZP[l], H, W, Ctot, Co4);
      const int gtotal = Npix * Co[l];
      gates_kernel<<<(gtotal + 255) / 256, 256, 0, stream>>>(
          ZP[l], CP[l], FP[l], LP[l], gg[l], be[l], mme[l], mva[l],
          H, W, Co[l], Cin[l], Ctot, t, (t == 0) ? 1 : 0, gtotal);
    }
  }

  // 3) final upsample into d_out: [2,8,128,128,32]
  const long ototal = (long)B_ * T_ * 128 * 128 * 32;
  upsample_out_kernel<<<(unsigned)((ototal + 255) / 256), 256, 0, stream>>>(
      LP[2], (float*)d_out, 64, 64, 32, ototal);
}
